// Model_3238405341467
// MI455X (gfx1250) — matrix operand order repaired and checked by reading, not compiled, not run on hardware
//
#include <hip/hip_runtime.h>
#include <hip/hip_fp16.h>

#define N_NODES 30000
#define N_EDGES 480000
#define HID     256
#define HEADS   8
#define DH      32
#define L_LAYERS 5

typedef __attribute__((ext_vector_type(16))) _Float16 v16h;
typedef __attribute__((ext_vector_type(8)))  _Float16 v8h;
typedef __attribute__((ext_vector_type(4)))  _Float16 v4h;
typedef __attribute__((ext_vector_type(8)))  float    v8f;
typedef __attribute__((ext_vector_type(4)))  float    v4f;
typedef __attribute__((ext_vector_type(4)))  int      v4i;
typedef __attribute__((ext_vector_type(8)))  int      v8i;

union U16 { v16h v; v8h h[2]; };

__device__ __forceinline__ float gelu_f(float x) {
    float x3 = x * x * x;
    return 0.5f * x * (1.0f + tanhf(0.7978845608028654f * (x + 0.044715f * x3)));
}

// ---------------------------------------------------------------------------
// WMMA GEMM: C[M x 256] = A[M x KD] * B[KD x 256] (+bias, epilogue by MODE)
// B is pre-transposed f16: Bt[n][k].
// A staging:  KD==256 f16  -> TDM tensor_load_to_lds (2D tile, LDS pad via D#)
//             KD==512 f16  -> global_load_async_to_lds_b128 per lane
//             A32 (f32)    -> VGPR staging with f32->f16 convert
// MODE: 0 = store f16, 1 = store f16 with gelu, 2 = store f32, 3 = f32 +=
// Block: 256 threads (8 waves) computes 64 rows x 256 cols.
// ---------------------------------------------------------------------------
template <int KD, int MODE, bool A32>
__global__ __launch_bounds__(256) void gemm_k(const void* __restrict__ Asrc,
                                              const _Float16* __restrict__ Bt,
                                              const float* __restrict__ bias,
                                              void* __restrict__ Out, int M) {
    constexpr int AST = 264;                 // 256 + 8 halves pad (conflict-free)
    __shared__ _Float16 As[64 * AST];
    const int tid  = threadIdx.x;
    const int wave = tid >> 5, lane = tid & 31;
    const int rbase = blockIdx.x * 64;
    const int rt = wave & 3;
    const int ch = wave >> 2;
    const int arow = rt * 16 + (lane & 15);
    const int kb = lane >> 4;
    const int bcol0 = ch * 128;

    v8f acc[8] = {};

    for (int kblk = 0; kblk < KD / 256; ++kblk) {
        __syncthreads();
        if (A32) {
            const float* A = (const float*)Asrc;
            for (int idx = tid; idx < 64 * 64; idx += 256) {
                int r = idx >> 6, c = (idx & 63) << 2;
                int gr = rbase + r;
                v4f a = {};
                if (gr < M) a = *(const v4f*)(A + (size_t)gr * KD + kblk * 256 + c);
                v4h t;
                t[0] = (_Float16)a[0]; t[1] = (_Float16)a[1];
                t[2] = (_Float16)a[2]; t[3] = (_Float16)a[3];
                *(v4h*)(As + r * AST + c) = t;
            }
        } else if (KD == 256) {
            // Tensor Data Mover: one 2D tile descriptor per workgroup.
            // tile = 64 rows x 256 halves, row stride KD halves, data_size=2B.
            // pad_enable: +4 DWORDs (8 halves) after every 128 DWORDs (one row)
            // -> reproduces the AST=264 padded LDS layout.  OOB rows read 0.
            if (tid == 0) {
                unsigned lds = (unsigned)(uintptr_t)(&As[0]);
                unsigned long long ga =
                    (unsigned long long)(uintptr_t)((const _Float16*)Asrc + (size_t)rbase * KD);
                int rows = M - rbase;
                if (rows > 64) rows = 64;
                v4i g0;
                g0[0] = 1;                                   // count=1, user D#
                g0[1] = (int)lds;                            // lds_addr
                g0[2] = (int)(unsigned)(ga & 0xFFFFFFFFull); // global_addr[31:0]
                g0[3] = (int)(((ga >> 32) & 0x01FFFFFFull) | 0x80000000u); // addr[56:32]|type=2
                v8i g1;
                g1[0] = 0x07910000;  // data_size=1(2B), pad_en, pad_interval=6, pad_amount=3
                g1[1] = (KD & 0xFFFF) << 16;                          // tensor_dim0[15:0]
                g1[2] = ((KD >> 16) & 0xFFFF) | ((rows & 0xFFFF) << 16); // td0[31:16]|td1[15:0]
                g1[3] = ((rows >> 16) & 0xFFFF) | (256 << 16);        // td1[31:16]|tile_dim0
                g1[4] = 64;                                           // tile_dim1 | tile_dim2=0
                g1[5] = KD;                                           // tensor_dim0_stride[31:0]
                g1[6] = 0;                                            // stride[47:32]|td1_stride lo
                g1[7] = 0;
                v4i gz; gz[0] = 0; gz[1] = 0; gz[2] = 0; gz[3] = 0;
                asm volatile("tensor_load_to_lds %0, %1, %2, %3"
                             :: "s"(g0), "s"(g1), "s"(gz), "s"(gz) : "memory");
                __builtin_amdgcn_s_wait_tensorcnt(0);
            }
        } else {
            // f16 source, KD=512: async global->LDS copy, no VGPR round-trip.
            const _Float16* A = (const _Float16*)Asrc;
            for (int idx = tid; idx < 64 * 32; idx += 256) {
                int r = idx >> 5, c = (idx & 31) << 3;
                int gr = rbase + r;
                _Float16* dst = As + r * AST + c;
                if (gr < M) {
                    unsigned ldsoff = (unsigned)(uintptr_t)dst;
                    unsigned long long ga =
                        (unsigned long long)(uintptr_t)(A + (size_t)gr * KD + kblk * 256 + c);
                    asm volatile("global_load_async_to_lds_b128 %0, %1, off"
                                 :: "v"(ldsoff), "v"(ga) : "memory");
                } else {
                    v8h z = {};
                    *(v8h*)dst = z;
                }
            }
            asm volatile("s_wait_asynccnt 0x0" ::: "memory");
        }
        __syncthreads();

#pragma unroll
        for (int ks = 0; ks < 8; ++ks) {
            // A fragment: 16-bit A 16x32 layout: lane group kb holds K [kb*8,kb*8+8)
            // in halves 0..7 and K [kb*8+16,kb*8+24) in halves 8..15.
            U16 a;
            const _Float16* ap = As + arow * AST + ks * 32 + kb * 8;
            a.h[0] = *(const v8h*)ap;
            a.h[1] = *(const v8h*)(ap + 16);
#pragma unroll
            for (int ct = 0; ct < 8; ++ct) {
                int col = bcol0 + ct * 16 + (lane & 15);
                // B fragment: lane group kb holds K [kb*8,kb*8+8) and [kb*8+16,kb*8+24).
                const _Float16* bp = Bt + (size_t)col * KD + kblk * 256 + ks * 32 + kb * 8;
                U16 b;
                b.h[0] = *(const v8h*)bp;
                b.h[1] = *(const v8h*)(bp + 16);
                acc[ct] = __builtin_amdgcn_wmma_f32_16x16x32_f16(
                    false, a.v, false, b.v, (short)0, acc[ct], false, false);
            }
        }
    }

#pragma unroll
    for (int ct = 0; ct < 8; ++ct) {
        int col = bcol0 + ct * 16 + (lane & 15);
        float bz = bias ? bias[col] : 0.0f;
#pragma unroll
        for (int j = 0; j < 8; ++j) {
            int row = rbase + rt * 16 + (lane >> 4) * 8 + j;
            if (row < M) {
                float v = acc[ct][j] + bz;
                if (MODE == 1) v = gelu_f(v);
                if (MODE == 0 || MODE == 1)
                    ((_Float16*)Out)[(size_t)row * HID + col] = (_Float16)v;
                else if (MODE == 2)
                    ((float*)Out)[(size_t)row * HID + col] = v;
                else
                    ((float*)Out)[(size_t)row * HID + col] += v;
            }
        }
    }
}

// ---------------------------------------------------------------------------
// LayerNorm over 256 cols; one wave per node; f32 in, f16 out.
// ---------------------------------------------------------------------------
__global__ __launch_bounds__(256) void ln_k(const float* __restrict__ h,
                                            const float* __restrict__ g,
                                            const float* __restrict__ b,
                                            _Float16* __restrict__ out, int n) {
    int widx = (blockIdx.x * 256 + threadIdx.x) >> 5;
    int lane = threadIdx.x & 31;
    if (widx >= n) return;
    const float* row = h + (size_t)widx * HID;
    v4f x0 = *(const v4f*)(row + lane * 8);
    v4f x1 = *(const v4f*)(row + lane * 8 + 4);
    float s = 0.f, ss = 0.f;
#pragma unroll
    for (int i = 0; i < 4; ++i) {
        s += x0[i] + x1[i];
        ss += x0[i] * x0[i] + x1[i] * x1[i];
    }
#pragma unroll
    for (int o = 16; o > 0; o >>= 1) {
        s  += __shfl_xor(s, o, 32);
        ss += __shfl_xor(ss, o, 32);
    }
    float mean = s * (1.0f / 256.0f);
    float var  = ss * (1.0f / 256.0f) - mean * mean;
    float rstd = rsqrtf(var + 1e-6f);
    _Float16* orow = out + (size_t)widx * HID;
#pragma unroll
    for (int i = 0; i < 8; ++i) {
        int c = lane * 8 + i;
        float x = (i < 4) ? x0[i] : x1[i - 4];
        orow[c] = (_Float16)((x - mean) * rstd * g[c] + b[c]);
    }
}

// preprocess stage 1: t = gelu(x @ W1 + b1), K=4, out f16 [N x 512]
__global__ void pre1_k(const float* __restrict__ x, const float* __restrict__ w1,
                       const float* __restrict__ b1, _Float16* __restrict__ t, int n) {
    int tid = blockIdx.x * 256 + threadIdx.x;
    if (tid >= n * 512) return;
    int node = tid >> 9, c = tid & 511;
    const float* xr = x + (size_t)node * 4;
    float a = xr[0] * w1[c] + xr[1] * w1[512 + c] + xr[2] * w1[1024 + c] +
              xr[3] * w1[1536 + c] + b1[c];
    t[tid] = (_Float16)gelu_f(a);
}

// transpose+convert W[K x 256] (f32) -> Bt[256 x K] (f16)
__global__ void wtrans_k(const float* __restrict__ w, _Float16* __restrict__ bt, int K) {
    int tid = blockIdx.x * 256 + threadIdx.x;
    if (tid >= K * 256) return;
    int k = tid >> 8, n = tid & 255;
    bt[(size_t)n * K + k] = (_Float16)w[(size_t)k * 256 + n];
}

__device__ __forceinline__ unsigned enc_f(float x) {
    unsigned u = __float_as_uint(x);
    return (u & 0x80000000u) ? ~u : (u | 0x80000000u);
}
__device__ __forceinline__ float dec_f(unsigned u) {
    unsigned b = (u & 0x80000000u) ? (u ^ 0x80000000u) : ~u;
    return __uint_as_float(b);
}

// edge pass A: per (edge, head) dot(q[r],k[s]) -> logit, atomicMax segment max
__global__ __launch_bounds__(256) void edgeA_k(const _Float16* __restrict__ q,
                                               const _Float16* __restrict__ k,
                                               const int* __restrict__ snd,
                                               const int* __restrict__ rcv,
                                               float* __restrict__ elog,
                                               unsigned* __restrict__ mmax) {
    int tid = blockIdx.x * 256 + threadIdx.x;
    if (tid >= N_EDGES * HEADS) return;
    int e = tid >> 3, hh = tid & 7;
    int s = snd[e], r = rcv[e];
    const _Float16* qp = q + (size_t)r * HID + hh * DH;
    const _Float16* kp = k + (size_t)s * HID + hh * DH;
    float acc = 0.f;
#pragma unroll
    for (int i = 0; i < 4; ++i) {
        v8h qv = *(const v8h*)(qp + i * 8);
        v8h kv = *(const v8h*)(kp + i * 8);
#pragma unroll
        for (int j = 0; j < 8; ++j) acc += (float)qv[j] * (float)kv[j];
    }
    acc *= 0.17677669529663687f;  // 1/sqrt(32)
    elog[tid] = acc;
    atomicMax(mmax + r * 8 + hh, enc_f(acc));
}

// edge pass B: ex = exp(logit - max); accumulate denom
__global__ __launch_bounds__(256) void edgeB_k(const int* __restrict__ rcv,
                                               float* __restrict__ elog,
                                               const unsigned* __restrict__ mmax,
                                               float* __restrict__ denom) {
    int tid = blockIdx.x * 256 + threadIdx.x;
    if (tid >= N_EDGES * HEADS) return;
    int e = tid >> 3, hh = tid & 7;
    int r = rcv[e];
    float m = dec_f(mmax[r * 8 + hh]);
    float ex = __expf(elog[tid] - m);
    elog[tid] = ex;
    atomicAdd(denom + r * 8 + hh, ex);
}

// edge pass C: one wave per edge; scatter alpha * v[s] into agg[r]
__global__ __launch_bounds__(256) void edgeC_k(const _Float16* __restrict__ v,
                                               const int* __restrict__ snd,
                                               const int* __restrict__ rcv,
                                               const float* __restrict__ elog,
                                               const float* __restrict__ denom,
                                               float* __restrict__ agg) {
    int e = (blockIdx.x * 256 + threadIdx.x) >> 5;
    int lane = threadIdx.x & 31;
    if (e >= N_EDGES) return;
    int s = snd[e], r = rcv[e];
    int hh = lane >> 2;
    float alpha = elog[e * 8 + hh] / (denom[r * 8 + hh] + 1e-9f);
    v8h vv = *(const v8h*)(v + (size_t)s * HID + lane * 8);
    float* ap = agg + (size_t)r * HID + lane * 8;
#pragma unroll
    for (int i = 0; i < 8; ++i) atomicAdd(ap + i, (float)vv[i] * alpha);
}

// final projection: out = h @ out_w[256x3] + out_b
__global__ void out_k(const float* __restrict__ h, const float* __restrict__ w,
                      const float* __restrict__ b, float* __restrict__ out) {
    int n = blockIdx.x * 256 + threadIdx.x;
    if (n >= N_NODES) return;
    const float* hr = h + (size_t)n * HID;
    float a0 = b[0], a1 = b[1], a2 = b[2];
    for (int c = 0; c < HID; ++c) {
        float x = hr[c];
        a0 += x * w[c * 3];
        a1 += x * w[c * 3 + 1];
        a2 += x * w[c * 3 + 2];
    }
    out[n * 3] = a0; out[n * 3 + 1] = a1; out[n * 3 + 2] = a2;
}

// ---------------------------------------------------------------------------
extern "C" void kernel_launch(void* const* d_in, const int* in_sizes, int n_in,
                              void* d_out, int out_size, void* d_ws, size_t ws_size,
                              hipStream_t stream) {
    (void)in_sizes; (void)n_in; (void)out_size; (void)ws_size;
    const float* x      = (const float*)d_in[0];
    const int*   snd    = (const int*)d_in[1];
    const int*   rcv    = (const int*)d_in[2];
    const float* pre_w1 = (const float*)d_in[3];
    const float* pre_b1 = (const float*)d_in[4];
    const float* pre_w2 = (const float*)d_in[5];
    const float* pre_b2 = (const float*)d_in[6];
    const float* ln1_g  = (const float*)d_in[7];
    const float* ln1_b  = (const float*)d_in[8];
    const float* ln2_g  = (const float*)d_in[9];
    const float* ln2_b  = (const float*)d_in[10];
    const float* wq = (const float*)d_in[11]; const float* bq = (const float*)d_in[12];
    const float* wk = (const float*)d_in[13]; const float* bk = (const float*)d_in[14];
    const float* wv = (const float*)d_in[15]; const float* bv = (const float*)d_in[16];
    const float* wo = (const float*)d_in[17]; const float* bo = (const float*)d_in[18];
    const float* mw1 = (const float*)d_in[19]; const float* mb1 = (const float*)d_in[20];
    const float* mw2 = (const float*)d_in[21]; const float* mb2 = (const float*)d_in[22];
    const float* out_w = (const float*)d_in[23]; const float* out_b = (const float*)d_in[24];

    char* ws = (char*)d_ws;
    size_t off = 0;
    auto alloc = [&](size_t bytes) -> void* {
        void* p = ws + off;
        off = (off + bytes + 255) & ~(size_t)255;
        return p;
    };
    const size_t N = N_NODES, E = N_EDGES;
    float*     h    = (float*)alloc(N * HID * 4);
    _Float16*  a16a = (_Float16*)alloc(N * 512 * 2);  // pre t16 / LN outputs
    _Float16*  a16b = (_Float16*)alloc(N * HID * 2);  // mlp hidden f16
    _Float16*  q16  = (_Float16*)alloc(N * HID * 2);
    _Float16*  k16  = (_Float16*)alloc(N * HID * 2);
    _Float16*  v16  = (_Float16*)alloc(N * HID * 2);
    float*     agg  = (float*)alloc(N * HID * 4);
    unsigned*  mmax = (unsigned*)alloc(N * HEADS * 4);
    float*     denom= (float*)alloc(N * HEADS * 4);
    float*     elog = (float*)alloc(E * HEADS * 4);
    _Float16*  wt   = (_Float16*)alloc((size_t)(256 * 512 + 30 * 256 * 256) * 2);

    _Float16* wt_pre2 = wt;                       // [256 x 512]
    auto wt_mat = [&](int layer, int m) -> _Float16* {
        return wt + 256 * 512 + (size_t)(layer * 6 + m) * 256 * 256;
    };

    const int GB = (N_NODES + 63) / 64;  // 469 gemm blocks

    // --- weight transpose/convert to f16 ---
    wtrans_k<<<512, 256, 0, stream>>>(pre_w2, wt_pre2, 512);
    for (int i = 0; i < L_LAYERS; ++i) {
        size_t wofs = (size_t)i * HID * HID;
        wtrans_k<<<256, 256, 0, stream>>>(wq  + wofs, wt_mat(i, 0), 256);
        wtrans_k<<<256, 256, 0, stream>>>(wk  + wofs, wt_mat(i, 1), 256);
        wtrans_k<<<256, 256, 0, stream>>>(wv  + wofs, wt_mat(i, 2), 256);
        wtrans_k<<<256, 256, 0, stream>>>(wo  + wofs, wt_mat(i, 3), 256);
        wtrans_k<<<256, 256, 0, stream>>>(mw1 + wofs, wt_mat(i, 4), 256);
        wtrans_k<<<256, 256, 0, stream>>>(mw2 + wofs, wt_mat(i, 5), 256);
    }

    // --- preprocess MLP ---
    pre1_k<<<(N_NODES * 512) / 256, 256, 0, stream>>>(x, pre_w1, pre_b1, a16a, N_NODES);
    gemm_k<512, 2, false><<<GB, 256, 0, stream>>>(a16a, wt_pre2, pre_b2, h, N_NODES);

    // --- transformer blocks ---
    for (int i = 0; i < L_LAYERS; ++i) {
        const size_t bofs = (size_t)i * HID;
        ln_k<<<3750, 256, 0, stream>>>(h, ln1_g + bofs, ln1_b + bofs, a16a, N_NODES);
        gemm_k<256, 0, false><<<GB, 256, 0, stream>>>(a16a, wt_mat(i, 0), bq + bofs, q16, N_NODES);
        gemm_k<256, 0, false><<<GB, 256, 0, stream>>>(a16a, wt_mat(i, 1), bk + bofs, k16, N_NODES);
        gemm_k<256, 0, false><<<GB, 256, 0, stream>>>(a16a, wt_mat(i, 2), bv + bofs, v16, N_NODES);

        hipMemsetAsync(mmax, 0, N * HEADS * 4, stream);
        hipMemsetAsync(denom, 0, N * HEADS * 4, stream);
        hipMemsetAsync(agg, 0, N * HID * 4, stream);
        edgeA_k<<<(N_EDGES * HEADS) / 256, 256, 0, stream>>>(q16, k16, snd, rcv, elog, mmax);
        edgeB_k<<<(N_EDGES * HEADS) / 256, 256, 0, stream>>>(rcv, elog, mmax, denom);
        edgeC_k<<<(N_EDGES * 32) / 256, 256, 0, stream>>>(v16, snd, rcv, elog, denom, agg);

        // h += agg @ wo + bo   (A operand is f32, converted during staging)
        gemm_k<256, 3, true><<<GB, 256, 0, stream>>>(agg, wt_mat(i, 3), bo + bofs, h, N_NODES);

        // MLP block
        ln_k<<<3750, 256, 0, stream>>>(h, ln2_g + bofs, ln2_b + bofs, a16a, N_NODES);
        gemm_k<256, 1, false><<<GB, 256, 0, stream>>>(a16a, wt_mat(i, 4), mb1 + bofs, a16b, N_NODES);
        gemm_k<256, 3, false><<<GB, 256, 0, stream>>>(a16b, wt_mat(i, 5), mb2 + bofs, h, N_NODES);
    }

    // --- output projection ---
    out_k<<<(N_NODES + 255) / 256, 256, 0, stream>>>(h, out_w, out_b, (float*)d_out);
}
